// Hierarchical_Fusion_86294482911529
// MI455X (gfx1250) — matrix operand order repaired and checked by reading, not compiled, not run on hardware
//
#include <hip/hip_runtime.h>
#include <cstdint>
#include <cstddef>

#define B_ 8
#define S_ 2048
#define M_ 2048
#define H_ 1024           // hidden == query dim
#define NW 8              // waves per attention block
#define HS 128            // hidden slice per wave
#define MC 128            // m-chunk (NW * 16)
#define QSTRIDE 1032      // padded LDS row pitch (ushorts) == TDM pad(interval=1024B, amount=8B)

typedef __attribute__((ext_vector_type(16))) __bf16 v16bf;
typedef __attribute__((ext_vector_type(8)))  float  v8f;
typedef __attribute__((ext_vector_type(4)))  unsigned int v4u;
typedef __attribute__((ext_vector_type(4)))  int v4i;
typedef __attribute__((ext_vector_type(8)))  int v8i;

union BfPack {
  v16bf v;
  uint4 q[2];
  unsigned short s[16];
};

__device__ __forceinline__ unsigned short f2bf(float f) {
  unsigned int x = __float_as_uint(f);
  x += 0x7FFFu + ((x >> 16) & 1u);   // round-to-nearest-even
  return (unsigned short)(x >> 16);
}

__device__ __forceinline__ v8f wmma_bf16(const BfPack& a, const BfPack& b, v8f c) {
  return __builtin_amdgcn_wmma_f32_16x16x32_bf16(
      /*neg_a=*/false, a.v, /*neg_b=*/false, b.v,
      /*c_mod=*/(short)0, c, /*reuse_a=*/false, /*reuse_b=*/false);
}

// TDM: load a 16x1024 bf16 tile (row stride 1024 elems) from global into LDS,
// with +8B LDS padding after every 1024B -> effective LDS row pitch 2064B (QSTRIDE).
__device__ __forceinline__ void tdm_load_qtile(const unsigned short* gsrc, unsigned lds_off) {
  unsigned long long ga = (unsigned long long)(uintptr_t)gsrc;
  v4u g0;
  g0.x = 1u;                                                   // count=1 (valid), load, user
  g0.y = lds_off;                                              // LDS byte address
  g0.z = (unsigned)(ga & 0xFFFFFFFFull);                       // global_addr[31:0]
  g0.w = (unsigned)((ga >> 32) & 0x1FFFFFFull) | (2u << 30);   // global_addr[56:32] | type=2
  v8i g1;
  g1[0] = (int)((1u << 16) | (1u << 20) | (7u << 22) | (1u << 25)); // data_size=2B, pad_en, interval=256dw, amount=2dw
  g1[1] = (int)(1024u << 16);                                  // tensor_dim0 = 1024
  g1[2] = (int)(16u << 16);                                    // tensor_dim1 = 16
  g1[3] = (int)(1024u << 16);                                  // tile_dim0 = 1024
  g1[4] = 16;                                                  // tile_dim1 = 16
  g1[5] = 1024;                                                // tensor_dim0_stride = 1024
  g1[6] = 0;
  g1[7] = 0;
  v4i z4 = {0, 0, 0, 0};
#if __clang_major__ >= 23
  v8i z8 = {0, 0, 0, 0, 0, 0, 0, 0};
  __builtin_amdgcn_tensor_load_to_lds(g0, g1, z4, z4, z8, 0);
#else
  __builtin_amdgcn_tensor_load_to_lds(g0, g1, z4, z4, 0);
#endif
}

// ---------------------------------------------------------------- generic f32 -> bf16 (8 elems/thread)
__global__ __launch_bounds__(256) void cvt_bf16_kernel(const float* __restrict__ src,
                                                       unsigned short* __restrict__ dst,
                                                       int n_oct) {
  int i = blockIdx.x * 256 + threadIdx.x;
  if (i >= n_oct) return;
  const float4* p = (const float4*)src;
  float4 a = p[2 * i], b = p[2 * i + 1];
  uint4 o;
  o.x = (unsigned)f2bf(a.x) | ((unsigned)f2bf(a.y) << 16);
  o.y = (unsigned)f2bf(a.z) | ((unsigned)f2bf(a.w) << 16);
  o.z = (unsigned)f2bf(b.x) | ((unsigned)f2bf(b.y) << 16);
  o.w = (unsigned)f2bf(b.z) | ((unsigned)f2bf(b.w) << 16);
  ((uint4*)dst)[i] = o;
}

// ---------------------------------------------------------------- info [B][M][H] f32 -> infoT [B][H][M] bf16
__global__ __launch_bounds__(256) void transpose_bf16_kernel(const float* __restrict__ info,
                                                             unsigned short* __restrict__ infoT) {
  __shared__ float tile[32][33];
  const int b  = blockIdx.z;
  const int m0 = blockIdx.x * 32;
  const int h0 = blockIdx.y * 32;
  const int tx = threadIdx.x & 31;
  const int ty = threadIdx.x >> 5;   // 0..7
  const float* src = info + ((size_t)b * M_ + m0) * H_ + h0;
#pragma unroll
  for (int i = 0; i < 4; ++i) {
    const int r = ty + i * 8;
    tile[r][tx] = src[(size_t)r * H_ + tx];
  }
  __syncthreads();
  unsigned short* dst = infoT + ((size_t)b * H_ + h0) * M_ + m0;
#pragma unroll
  for (int i = 0; i < 4; ++i) {
    const int r = ty + i * 8;
    dst[(size_t)r * M_ + tx] = f2bf(tile[tx][r]);
  }
}

// ---------------------------------------------------------------- Q projection: q = query @ W^T + b (bf16 out)
__global__ __launch_bounds__(128) void proj_kernel(const unsigned short* __restrict__ Qin,
                                                   const unsigned short* __restrict__ Wb,
                                                   const float* __restrict__ bias,
                                                   unsigned short* __restrict__ Qbf) {
  const int lane = threadIdx.x & 31;
  const int wave = threadIdx.x >> 5;
  const int l15  = lane & 15;
  const int half = lane >> 4;
  const int rowbase = blockIdx.x * 16;                 // flattened b*S + s
  const int nbase   = blockIdx.y * 256 + wave * 64;

  const unsigned short* qrow = Qin + (size_t)(rowbase + l15) * H_;
  v8f acc[4];
#pragma unroll
  for (int t = 0; t < 4; ++t)
#pragma unroll
    for (int e = 0; e < 8; ++e) acc[t][e] = 0.0f;

#pragma unroll 2
  for (int k = 0; k < H_; k += 32) {
    BfPack A;
    const unsigned short* ap = qrow + k + half * 8;
    A.q[0] = *(const uint4*)(ap);
    A.q[1] = *(const uint4*)(ap + 16);
#pragma unroll
    for (int t = 0; t < 4; ++t) {
      const int ncol = nbase + t * 16 + l15;
      const unsigned short* bp = Wb + (size_t)ncol * H_ + k + half * 8;
      BfPack Bm;
      Bm.q[0] = *(const uint4*)(bp);
      Bm.q[1] = *(const uint4*)(bp + 16);
      acc[t] = wmma_bf16(A, Bm, acc[t]);
    }
  }

#pragma unroll
  for (int t = 0; t < 4; ++t) {
    const int ncol = nbase + t * 16 + l15;
    const float bb = bias[ncol];
#pragma unroll
    for (int e = 0; e < 8; ++e) {
      const int row = rowbase + e + 8 * half;     // C layout: lanes 0-15 -> M=e, 16-31 -> M=e+8
      Qbf[(size_t)row * H_ + ncol] = f2bf(acc[t][e] + bb);
    }
  }
}

// ---------------------------------------------------------------- fused flash attention
__global__ __launch_bounds__(256) void attn_kernel(const unsigned short* __restrict__ Qbf,
                                                   const unsigned short* __restrict__ infoB,
                                                   const unsigned short* __restrict__ infoT,
                                                   float* __restrict__ out) {
  __shared__ unsigned short sQ[16 * QSTRIDE];     // staged Q rows (bf16), TDM-padded pitch
  __shared__ unsigned short sP[16][MC];           // P tile (bf16), rows x m-cols
  __shared__ float sRed[NW][16];
  __shared__ float sSum[NW][16];
  __shared__ __align__(16) float sM[16];
  __shared__ __align__(16) float sL[16];
  __shared__ __align__(16) float sScale[16];

  const int lane = threadIdx.x & 31;
  const int wave = threadIdx.x >> 5;
  const int l15  = lane & 15;
  const int half = lane >> 4;
  const int hbase = wave * HS;
  const int rowbase = blockIdx.x * 16;            // flattened b*S + s
  const int bidx = blockIdx.x >> 7;               // / (S_/16)

  const unsigned short* qrows = Qbf + (size_t)rowbase * H_;
  const unsigned short* ib    = infoB + (size_t)bidx * M_ * H_;   // [m][h] row-major
  const unsigned short* iT    = infoT + (size_t)bidx * H_ * M_;   // [h][m] row-major

  // ---- stage the block's 16 Q rows into LDS via the Tensor Data Mover (wave 0 only)
  if (wave == 0) {
    tdm_load_qtile(qrows, (unsigned)(uintptr_t)&sQ[0]);
    __builtin_amdgcn_s_wait_tensorcnt(0);
  }

  v8f O[8];
#pragma unroll
  for (int t = 0; t < 8; ++t)
#pragma unroll
    for (int e = 0; e < 8; ++e) O[t][e] = 0.0f;

  if (threadIdx.x < 16) { sM[threadIdx.x] = -3.0e38f; sL[threadIdx.x] = 0.0f; }
  __syncthreads();

  for (int m0 = 0; m0 < M_; m0 += MC) {
    // ---- scores tile: this wave computes S[16 rows x 16 cols] at mw
    const int mw = m0 + wave * 16;
    v8f sc;
#pragma unroll
    for (int e = 0; e < 8; ++e) sc[e] = 0.0f;

    if (m0 + MC < M_) {   // warm L2 for the next chunk (global_prefetch_b8)
      __builtin_prefetch(ib + (size_t)(mw + MC + l15) * H_, 0, 0);
      __builtin_prefetch(iT + (size_t)(hbase + l15) * M_ + m0 + MC, 0, 0);
    }

#pragma unroll 4
    for (int k = 0; k < H_; k += 32) {
      BfPack A, Bm;
      const int kk = k + ((k >> 9) << 2);   // +4 ushorts past col 512 (TDM pad shift)
      const unsigned short* ap = &sQ[l15 * QSTRIDE + kk + half * 8];
      A.q[0] = *(const uint4*)(ap);
      A.q[1] = *(const uint4*)(ap + 16);
      const unsigned short* bp = ib + (size_t)(mw + l15) * H_ + k + half * 8;
      Bm.q[0] = *(const uint4*)(bp);
      Bm.q[1] = *(const uint4*)(bp + 16);
      sc = wmma_bf16(A, Bm, sc);
    }

    // ---- per-row max within tile (reduce across 16 lanes)
    {
      v8f r = sc;
#pragma unroll
      for (int off = 8; off; off >>= 1)
#pragma unroll
        for (int e = 0; e < 8; ++e) r[e] = fmaxf(r[e], __shfl_xor(r[e], off, 32));
      if (l15 == 0) {
#pragma unroll
        for (int e = 0; e < 8; ++e) sRed[wave][e + 8 * half] = r[e];
      }
    }
    __syncthreads();

    // ---- combine maxes across waves, update running max + scale
    if (threadIdx.x < 16) {
      float t = sRed[0][threadIdx.x];
#pragma unroll
      for (int w = 1; w < NW; ++w) t = fmaxf(t, sRed[w][threadIdx.x]);
      const float mo = sM[threadIdx.x];
      const float mn = fmaxf(mo, t);
      sScale[threadIdx.x] = __expf(mo - mn);
      sM[threadIdx.x] = mn;
    }
    __syncthreads();

    // ---- P = exp(S - m_new), row sums, publish P (bf16) to LDS
    {
      float4 mn0 = *(const float4*)&sM[8 * half];
      float4 mn1 = *(const float4*)&sM[8 * half + 4];
      float mn[8] = {mn0.x, mn0.y, mn0.z, mn0.w, mn1.x, mn1.y, mn1.z, mn1.w};
      v8f P;
#pragma unroll
      for (int e = 0; e < 8; ++e) P[e] = __expf(sc[e] - mn[e]);
      v8f rs = P;
#pragma unroll
      for (int off = 8; off; off >>= 1)
#pragma unroll
        for (int e = 0; e < 8; ++e) rs[e] += __shfl_xor(rs[e], off, 32);
      if (l15 == 0) {
#pragma unroll
        for (int e = 0; e < 8; ++e) sSum[wave][e + 8 * half] = rs[e];
      }
#pragma unroll
      for (int e = 0; e < 8; ++e) sP[e + 8 * half][wave * 16 + l15] = f2bf(P[e]);
    }
    __syncthreads();

    // ---- update running l; scale O accumulators by exp(m_old - m_new)
    if (threadIdx.x < 16) {
      float s = 0.0f;
#pragma unroll
      for (int w = 0; w < NW; ++w) s += sSum[w][threadIdx.x];
      sL[threadIdx.x] = sL[threadIdx.x] * sScale[threadIdx.x] + s;
    }
    {
      float4 s0 = *(const float4*)&sScale[8 * half];
      float4 s1 = *(const float4*)&sScale[8 * half + 4];
      float sv[8] = {s0.x, s0.y, s0.z, s0.w, s1.x, s1.y, s1.z, s1.w};
#pragma unroll
      for (int t = 0; t < 8; ++t)
#pragma unroll
        for (int e = 0; e < 8; ++e) O[t][e] *= sv[e];
    }

    // ---- O += P @ info[m0:m0+128, hslice]: A from LDS (P), B from pre-transposed infoT
#pragma unroll
    for (int kb = 0; kb < 4; ++kb) {
      BfPack A;   // P rows: lane row = l15, K = kb*32 + half-dependent chunks
      const unsigned short* ap = &sP[l15][kb * 32 + half * 8];
      A.q[0] = *(const uint4*)(ap);
      A.q[1] = *(const uint4*)(ap + 16);
#pragma unroll
      for (int nt = 0; nt < 8; ++nt) {
        const int hcol = hbase + nt * 16 + l15;
        const unsigned short* bp = iT + (size_t)hcol * M_ + m0 + kb * 32 + half * 8;
        BfPack Bm;
        Bm.q[0] = *(const uint4*)(bp);
        Bm.q[1] = *(const uint4*)(bp + 16);
        O[nt] = wmma_bf16(A, Bm, O[nt]);
      }
    }
    __syncthreads();   // sP reuse guard for next m-chunk
  }

  // ---- finalize: divide by l, store f32
  {
    float4 l0 = *(const float4*)&sL[8 * half];
    float4 l1 = *(const float4*)&sL[8 * half + 4];
    float inv[8] = {1.0f / l0.x, 1.0f / l0.y, 1.0f / l0.z, 1.0f / l0.w,
                    1.0f / l1.x, 1.0f / l1.y, 1.0f / l1.z, 1.0f / l1.w};
#pragma unroll
    for (int nt = 0; nt < 8; ++nt) {
      const int hcol = hbase + nt * 16 + l15;
#pragma unroll
      for (int e = 0; e < 8; ++e) {
        const int row = rowbase + e + 8 * half;
        out[(size_t)row * H_ + hcol] = O[nt][e] * inv[e];
      }
    }
  }
}

// ----------------------------------------------------------------
extern "C" void kernel_launch(void* const* d_in, const int* in_sizes, int n_in,
                              void* d_out, int out_size, void* d_ws, size_t ws_size,
                              hipStream_t stream) {
  (void)in_sizes; (void)n_in; (void)out_size; (void)ws_size;
  const float* query = (const float*)d_in[0];
  const float* info  = (const float*)d_in[1];
  const float* W     = (const float*)d_in[2];
  const float* bias  = (const float*)d_in[3];
  float* out = (float*)d_out;

  // workspace layout (bf16 buffers)
  unsigned short* Qbf   = (unsigned short*)d_ws;                 // projected q       [B*S][H]
  unsigned short* infoB = Qbf   + (size_t)B_ * S_ * H_;          // info bf16         [B][M][H]
  unsigned short* infoT = infoB + (size_t)B_ * M_ * H_;          // info^T bf16       [B][H][M]
  unsigned short* Qin   = infoT + (size_t)B_ * H_ * M_;          // query bf16        [B*S][H]
  unsigned short* Wb    = Qin   + (size_t)B_ * S_ * H_;          // W bf16            [H][H]

  const int oct_info  = B_ * M_ * H_ / 8;
  const int oct_query = B_ * S_ * H_ / 8;
  const int oct_w     = H_ * H_ / 8;
  cvt_bf16_kernel<<<(oct_info  + 255) / 256, 256, 0, stream>>>(info,  infoB, oct_info);
  cvt_bf16_kernel<<<(oct_query + 255) / 256, 256, 0, stream>>>(query, Qin,   oct_query);
  cvt_bf16_kernel<<<(oct_w     + 255) / 256, 256, 0, stream>>>(W,     Wb,    oct_w);
  transpose_bf16_kernel<<<dim3(M_ / 32, H_ / 32, B_), 256, 0, stream>>>(info, infoT);

  proj_kernel<<<dim3(B_ * S_ / 16, H_ / 256), 128, 0, stream>>>(Qin, Wb, bias, Qbf);
  attn_kernel<<<B_ * S_ / 16, 256, 0, stream>>>(Qbf, infoB, infoT, out);
}
